// VQTokenizer_34995393527977
// MI455X (gfx1250) — matrix operand order repaired and checked by reading, not compiled, not run on hardware
//
#include <hip/hip_runtime.h>

// CDNA5 / gfx1250 VQ encode: d2 = ||c||^2 - 2*<x,c> (+||x||^2 const per row),
// argmin over K via v_wmma_f32_16x16x32_f16, then exact f32 gather.
// Each wave owns 32 x-rows (two B-fragment sets) so every A-fragment pair
// staged through LDS feeds 4 WMMAs.

typedef __attribute__((ext_vector_type(16))) _Float16 v16h;
typedef __attribute__((ext_vector_type(8)))  float    v8f;

#define VQ_D      64     // dims
#define VQ_CHUNK  256    // codebook rows staged in LDS per iteration
#define VQ_STRIDE 72     // padded f16 row stride in LDS (144 B, bank spread)

__device__ __forceinline__ unsigned pack_h2(float a, float b) {
  _Float16 ha = (_Float16)a, hb2 = (_Float16)b;
  unsigned short ua, ub;
  __builtin_memcpy(&ua, &ha, 2);
  __builtin_memcpy(&ub, &hb2, 2);
  return (unsigned)ua | ((unsigned)ub << 16);
}

// B-matrix fragment (32x16 K-major): this lane holds 16 consecutive K values
// (K base chosen by caller) of its column; load 16 consecutive f32, cvt to f16.
__device__ __forceinline__ v16h load_b_frag(const float* p) {
  const float4* q = (const float4*)p;
  float4 f0 = q[0], f1 = q[1], f2 = q[2], f3 = q[3];
  v16h v;
  v[0]  = (_Float16)f0.x; v[1]  = (_Float16)f0.y; v[2]  = (_Float16)f0.z; v[3]  = (_Float16)f0.w;
  v[4]  = (_Float16)f1.x; v[5]  = (_Float16)f1.y; v[6]  = (_Float16)f1.z; v[7]  = (_Float16)f1.w;
  v[8]  = (_Float16)f2.x; v[9]  = (_Float16)f2.y; v[10] = (_Float16)f2.z; v[11] = (_Float16)f2.w;
  v[12] = (_Float16)f3.x; v[13] = (_Float16)f3.y; v[14] = (_Float16)f3.z; v[15] = (_Float16)f3.w;
  return v;
}

// A-matrix fragment (16x32, 16-bit): elems 0..15 = K{koff..koff+15}
// of this lane's row (contiguous order, matching B).
__device__ __forceinline__ v16h load_a_frag(const _Float16* base, int koff) {
  union { v16h v; uint4 q[2]; } u;
  u.q[0] = *(const uint4*)(base + koff);
  u.q[1] = *(const uint4*)(base + koff + 8);
  return u.v;
}

struct Best { float v; int i; };

__device__ __forceinline__ void upd8(Best& b, const v8f& acc,
                                     float4 c2a, float4 c2b, int cbase) {
  float s0 = c2a.x - 2.f * acc[0];
  float s1 = c2a.y - 2.f * acc[1];
  float s2 = c2a.z - 2.f * acc[2];
  float s3 = c2a.w - 2.f * acc[3];
  float s4 = c2b.x - 2.f * acc[4];
  float s5 = c2b.y - 2.f * acc[5];
  float s6 = c2b.z - 2.f * acc[6];
  float s7 = c2b.w - 2.f * acc[7];
  if (s0 < b.v) { b.v = s0; b.i = cbase + 0; }
  if (s1 < b.v) { b.v = s1; b.i = cbase + 1; }
  if (s2 < b.v) { b.v = s2; b.i = cbase + 2; }
  if (s3 < b.v) { b.v = s3; b.i = cbase + 3; }
  if (s4 < b.v) { b.v = s4; b.i = cbase + 4; }
  if (s5 < b.v) { b.v = s5; b.i = cbase + 5; }
  if (s6 < b.v) { b.v = s6; b.i = cbase + 6; }
  if (s7 < b.v) { b.v = s7; b.i = cbase + 7; }
}

__global__ __launch_bounds__(256) void vq_encode_kernel(
    const float* __restrict__ x, const float* __restrict__ cb,
    int* __restrict__ enc, float* __restrict__ quant, int N, int K)
{
  __shared__ __align__(16) _Float16 s_cb[VQ_CHUNK * VQ_STRIDE]; // ~36 KB
  __shared__ __align__(16) float    s_c2[VQ_CHUNK];             // 1 KB

  const int tid  = threadIdx.x;
  const int lane = tid & 31;
  const int wave = tid >> 5;
  const int hb   = lane >> 4;      // half-wave select (K/M split)
  const int l16  = lane & 15;
  const long rowbase = (long)blockIdx.x * 256 + (long)wave * 32;
  const long row0    = rowbase + l16;        // B set 0: rows rowbase+0..15
  const long row1    = row0 + 16;            // B set 1: rows rowbase+16..31

  // x rows as WMMA B fragments (in registers for the whole kernel).
  v16h b0a = load_b_frag(x + row0 * VQ_D +      hb * 16);  // set0 dims [0,32)
  v16h b0b = load_b_frag(x + row0 * VQ_D + 32 + hb * 16);  // set0 dims [32,64)
  v16h b1a = load_b_frag(x + row1 * VQ_D +      hb * 16);  // set1 dims [0,32)
  v16h b1b = load_b_frag(x + row1 * VQ_D + 32 + hb * 16);  // set1 dims [32,64)

  Best best0 = { 3.402823466e+38f, 0 };
  Best best1 = { 3.402823466e+38f, 0 };
  const int moff    = hb * 8;          // which 8 code-rows this lane's D holds
  const int nchunks = K / VQ_CHUNK;

  for (int chunk = 0; chunk < nchunks; ++chunk) {
    __syncthreads();
    {
      // Cooperative stage: thread t converts codebook row (chunk*256 + t)
      // to f16 in LDS and computes its exact f32 squared norm.
      const float* cr = cb + (long)(chunk * VQ_CHUNK + tid) * VQ_D;
      _Float16* dst = s_cb + tid * VQ_STRIDE;
      float sum = 0.f;
      #pragma unroll
      for (int i = 0; i < 8; ++i) {
        float4 f0 = ((const float4*)cr)[2 * i];
        float4 f1 = ((const float4*)cr)[2 * i + 1];
        sum += f0.x * f0.x + f0.y * f0.y + f0.z * f0.z + f0.w * f0.w
             + f1.x * f1.x + f1.y * f1.y + f1.z * f1.z + f1.w * f1.w;
        uint4 q;
        q.x = pack_h2(f0.x, f0.y);
        q.y = pack_h2(f0.z, f0.w);
        q.z = pack_h2(f1.x, f1.y);
        q.w = pack_h2(f1.z, f1.w);
        *(uint4*)(dst + i * 8) = q;
      }
      s_c2[tid] = sum;
      if (chunk + 1 < nchunks)  // warm L2 for next chunk (global_prefetch_b8)
        __builtin_prefetch(cb + (long)((chunk + 1) * VQ_CHUNK + tid) * VQ_D, 0, 0);
    }
    __syncthreads();

    #pragma unroll 4
    for (int t = 0; t < VQ_CHUNK / 16; ++t) {
      const _Float16* arow = s_cb + (t * 16 + l16) * VQ_STRIDE;
      v16h a0 = load_a_frag(arow,      hb * 16);   // dims [0,32)
      v16h a1 = load_a_frag(arow + 32, hb * 16);   // dims [32,64)

      // Each A-fragment pair feeds 4 WMMAs (two independent chains).
      v8f acc0 = {};
      v8f acc1 = {};
      acc0 = __builtin_amdgcn_wmma_f32_16x16x32_f16(false, a0, false, b0a,
                                                    (short)0, acc0, false, false);
      acc1 = __builtin_amdgcn_wmma_f32_16x16x32_f16(false, a0, false, b1a,
                                                    (short)0, acc1, false, false);
      acc0 = __builtin_amdgcn_wmma_f32_16x16x32_f16(false, a1, false, b0b,
                                                    (short)0, acc0, false, false);
      acc1 = __builtin_amdgcn_wmma_f32_16x16x32_f16(false, a1, false, b1b,
                                                    (short)0, acc1, false, false);

      // Shared c2 tile for both row sets (lane holds codes cbase+0..7).
      const float4* c2p = (const float4*)(s_c2 + t * 16 + moff);
      float4 c2a = c2p[0];
      float4 c2b = c2p[1];
      const int cbase = chunk * VQ_CHUNK + t * 16 + moff;
      upd8(best0, acc0, c2a, c2b, cbase);
      upd8(best1, acc1, c2a, c2b, cbase);
    }
  }

  // Merge lane L (codes m%16 in 0..7) with lane L+16 (codes 8..15); tie -> lower idx.
  {
    float ov = __shfl_xor(best0.v, 16, 32);
    int   oi = __shfl_xor(best0.i, 16, 32);
    if (ov < best0.v || (ov == best0.v && oi < best0.i)) { best0.v = ov; best0.i = oi; }
    ov = __shfl_xor(best1.v, 16, 32);
    oi = __shfl_xor(best1.i, 16, 32);
    if (ov < best1.v || (ov == best1.v && oi < best1.i)) { best1.v = ov; best1.i = oi; }
  }

  if (hb == 0) {                 // lanes 0..15 own the results
    enc[row0] = best0.i;
    enc[row1] = best1.i;
  }

  // Quantized gather: exact f32 codebook rows; 32 rows per wave, 2 floats/lane.
  #pragma unroll 1
  for (int r = 0; r < 16; ++r) {
    int idx0 = __shfl(best0.i, r, 32);
    int idx1 = __shfl(best1.i, r, 32);
    const float2* src0 = (const float2*)(cb + (long)idx0 * VQ_D);
    const float2* src1 = (const float2*)(cb + (long)idx1 * VQ_D);
    float2* dst0 = (float2*)(quant + (rowbase + r) * VQ_D);
    float2* dst1 = (float2*)(quant + (rowbase + 16 + r) * VQ_D);
    dst0[lane] = src0[lane];
    dst1[lane] = src1[lane];
  }
}

extern "C" void kernel_launch(void* const* d_in, const int* in_sizes, int n_in,
                              void* d_out, int out_size, void* d_ws, size_t ws_size,
                              hipStream_t stream) {
  const float* x  = (const float*)d_in[0];   // (N, 64) f32
  const float* cb = (const float*)d_in[1];   // (K, 64) f32
  const int N = in_sizes[0] / VQ_D;          // 262144
  const int K = in_sizes[1] / VQ_D;          // 1024
  int*   enc   = (int*)d_out;                // encoded (int32 bits), N slots
  float* quant = (float*)d_out + N;          // quantized, N*64 f32
  dim3 grid(N / 256), block(256);
  vq_encode_kernel<<<grid, block, 0, stream>>>(x, cb, enc, quant, N, K);
}
